// GNNResBlock_46849503264902
// MI455X (gfx1250) — matrix operand order repaired and checked by reading, not compiled, not run on hardware
//
#include <hip/hip_runtime.h>
#include <cmath>

typedef _Float16 h16;
typedef __attribute__((ext_vector_type(16))) _Float16 v16h;
typedef __attribute__((ext_vector_type(8)))  _Float16 v8h;
typedef __attribute__((ext_vector_type(8)))  float    v8f;

#define WMMA_F16(a, b, c) \
  __builtin_amdgcn_wmma_f32_16x16x32_f16(false, (a), false, (b), (short)0, (c), false, false)

__device__ __forceinline__ float silu_f(float v) {
  return v * (1.0f / (1.0f + __expf(-v)));
}

// A-fragment (16x32 f16, M x K) from an LDS tile with row stride `stride` halves.
// Per ISA 7.12.2: lanes 0-15 hold K 0..7 & 16..23; lanes 16-31 hold K 8..15 & 24..31.
__device__ __forceinline__ v16h load_a_frag(const h16* Abase, int stride, int lane, int kk) {
  int m    = lane & 15;
  int koff = kk * 32 + ((lane & 16) ? 8 : 0);
  const h16* p = Abase + m * stride + koff;
  v8h lo = *(const v8h*)(p);
  v8h hi = *(const v8h*)(p + 16);
  v16h r;
#pragma unroll
  for (int i = 0; i < 8; ++i) { r[i] = lo[i]; r[i + 8] = hi[i]; }
  return r;
}

// B-fragment (32x16 f16, K x N) from N-major (transposed) weights, row stride Kstride halves.
// Lanes 0-15 hold K 0..7 & 16..23 of column n; lanes 16-31 hold K 8..15 & 24..31 (matches A).
__device__ __forceinline__ v16h load_b_frag(const h16* Wt, int Kstride, int lane, int tile, int kk) {
  int n  = tile * 16 + (lane & 15);
  int kb = kk * 32 + ((lane & 16) ? 8 : 0);
  const h16* p = Wt + (size_t)n * Kstride + kb;
  v8h lo = *(const v8h*)(p);
  v8h hi = *(const v8h*)(p + 16);
  v16h r;
#pragma unroll
  for (int i = 0; i < 8; ++i) { r[i] = lo[i]; r[i + 8] = hi[i]; }
  return r;
}

// ---------------- utility kernels ----------------

__global__ __launch_bounds__(256) void zero_kernel(float* __restrict__ p, int n) {
  int i = blockIdx.x * 256 + threadIdx.x;
  if (i < n) p[i] = 0.0f;
}

// dst[n*K + k] = (f16) src[k*Nw + n]   (K x Nw fp32 -> Nw x K f16)
__global__ __launch_bounds__(256) void transpose_f16_kernel(const float* __restrict__ src,
                                                            h16* __restrict__ dst,
                                                            int K, int Nw) {
  int idx = blockIdx.x * 256 + threadIdx.x;
  if (idx >= K * Nw) return;
  int k = idx / Nw, n = idx % Nw;
  dst[(size_t)n * K + k] = (h16)src[(size_t)k * Nw + n];
}

// One row (C=128) per block; LayerNorm, write optional fp32 + f16 outputs.
__global__ __launch_bounds__(128) void ln_kernel(const float* __restrict__ in,
                                                 const float* __restrict__ g,
                                                 const float* __restrict__ b,
                                                 float* __restrict__ o32,
                                                 h16* __restrict__ o16) {
  __shared__ float s1[128];
  __shared__ float s2[128];
  int t = threadIdx.x;
  size_t rb = (size_t)blockIdx.x * 128;
  float v = in[rb + t];
  s1[t] = v;
  s2[t] = v * v;
  __syncthreads();
  for (int off = 64; off > 0; off >>= 1) {
    if (t < off) { s1[t] += s1[t + off]; s2[t] += s2[t + off]; }
    __syncthreads();
  }
  float mu  = s1[0] * (1.0f / 128.0f);
  float var = s2[0] * (1.0f / 128.0f) - mu * mu;
  float rs  = rsqrtf(var + 1e-5f);
  float y   = (v - mu) * rs * g[t] + b[t];
  if (o32) o32[rb + t] = y;
  o16[rb + t] = (h16)y;
}

// ---------------- fused edge MLP + cutoff + scatter ----------------
// 64 edges per block (4 M-groups of 16), 256 threads = 8 waves.
// Each wave owns 2 N-tiles; B-fragments are register-resident and reused across
// the 4 M-groups (8 WMMAs per B-load pair) -> 4x less weight traffic than M=16.
// Layer-1 accumulators stay in VGPRs so one LDS tile is reused for both layers.
__global__ __launch_bounds__(256) void edge_mlp_kernel(
    const float* __restrict__ x, const int* __restrict__ ei, int E,
    const h16* __restrict__ hnh,
    const h16* __restrict__ W1t, const float* __restrict__ W1last,
    const float* __restrict__ be1,
    const h16* __restrict__ W2t, const float* __restrict__ be2,
    float* __restrict__ sums, float* __restrict__ cnt) {
  __shared__ h16   Abuf[64 * 264];   // layer-1 A, then overwritten with layer-2 A
  __shared__ float s_dist[64];
  __shared__ float s_ccut[64];
  __shared__ int   s_col[64];

  const int tid   = threadIdx.x;
  const int ebase = blockIdx.x * 64;

  if (tid < 64) {
    int e = ebase + tid;
    int r = ei[e];
    int c = ei[E + e];
    float dx = x[3 * r + 0] - x[3 * c + 0];
    float dy = x[3 * r + 1] - x[3 * c + 1];
    float dz = x[3 * r + 2] - x[3 * c + 2];
    float d  = sqrtf(dx * dx + dy * dy + dz * dz);
    float cc = 0.5f * (__cosf(d * 0.62831853071795864769f) + 1.0f);
    cc = (d <= 5.0f) ? cc : 0.0f;
    s_dist[tid] = d;
    s_ccut[tid] = cc;
    s_col[tid]  = c;
    atomicAdd(&cnt[c], 1.0f);
  }
  // cooperative gather A = [hn[row] | hn[col]] (64 rows x 256 halves)
#pragma unroll
  for (int it = 0; it < 4; ++it) {
    int idx = tid + it * 256;    // 0..1023 chunk id
    int m   = idx >> 4;          // edge row 0..63
    int c16 = idx & 15;          // chunk of 16 halves
    int e   = ebase + m;
    int node = (c16 < 8) ? ei[e] : ei[E + e];
    int koff = (c16 < 8) ? c16 * 16 : (c16 - 8) * 16;
    const h16* src = hnh + (size_t)node * 128 + koff;
    v8h a0 = *(const v8h*)(src);
    v8h a1 = *(const v8h*)(src + 8);
    *(v8h*)&Abuf[m * 264 + c16 * 16]     = a0;
    *(v8h*)&Abuf[m * 264 + c16 * 16 + 8] = a1;
  }
  __syncthreads();

  const int lane   = tid & 31;
  const int wave   = tid >> 5;
  const int rowoff = (lane & 16) ? 8 : 0;
  const int ncl    = lane & 15;
  const int tile0  = wave * 2;
  const int tile1  = wave * 2 + 1;
  const int n0     = tile0 * 16 + ncl;
  const int n1     = tile1 * 16 + ncl;

  v8f acc[2][4];

  // ----- layer 1: (64x257) @ We1 -> 64x256 ; dist column + bias in acc init -----
  {
    float b00 = be1[n0], wl0 = W1last[n0];
    float b01 = be1[n1], wl1 = W1last[n1];
#pragma unroll
    for (int g = 0; g < 4; ++g)
#pragma unroll
      for (int r = 0; r < 8; ++r) {
        float d = s_dist[g * 16 + r + rowoff];
        acc[0][g][r] = b00 + d * wl0;
        acc[1][g][r] = b01 + d * wl1;
      }
  }
#pragma unroll
  for (int kk = 0; kk < 8; ++kk) {
    v16h b0 = load_b_frag(W1t, 256, lane, tile0, kk);
    v16h b1 = load_b_frag(W1t, 256, lane, tile1, kk);
#pragma unroll
    for (int g = 0; g < 4; ++g) {
      v16h a = load_a_frag(Abuf + g * 16 * 264, 264, lane, kk);
      acc[0][g] = WMMA_F16(a, b0, acc[0][g]);
      acc[1][g] = WMMA_F16(a, b1, acc[1][g]);
    }
  }
  __syncthreads();  // all waves done reading layer-1 A

  // SiLU -> f16, overwrite Abuf with layer-2 activations (64 x 256)
#pragma unroll
  for (int g = 0; g < 4; ++g)
#pragma unroll
    for (int r = 0; r < 8; ++r) {
      int m = g * 16 + r + rowoff;
      Abuf[m * 264 + n0] = (h16)silu_f(acc[0][g][r]);
      Abuf[m * 264 + n1] = (h16)silu_f(acc[1][g][r]);
    }
  __syncthreads();

  // ----- layer 2 + cosine cutoff + segment-sum scatter -----
  {
    float b00 = be2[n0];
    float b01 = be2[n1];
#pragma unroll
    for (int g = 0; g < 4; ++g)
#pragma unroll
      for (int r = 0; r < 8; ++r) {
        acc[0][g][r] = b00;
        acc[1][g][r] = b01;
      }
  }
#pragma unroll
  for (int kk = 0; kk < 8; ++kk) {
    v16h b0 = load_b_frag(W2t, 256, lane, tile0, kk);
    v16h b1 = load_b_frag(W2t, 256, lane, tile1, kk);
#pragma unroll
    for (int g = 0; g < 4; ++g) {
      v16h a = load_a_frag(Abuf + g * 16 * 264, 264, lane, kk);
      acc[0][g] = WMMA_F16(a, b0, acc[0][g]);
      acc[1][g] = WMMA_F16(a, b1, acc[1][g]);
    }
  }
#pragma unroll
  for (int g = 0; g < 4; ++g)
#pragma unroll
    for (int r = 0; r < 8; ++r) {
      int m = g * 16 + r + rowoff;
      float cc = s_ccut[m];
      size_t srow = (size_t)s_col[m] * 256;
      atomicAdd(&sums[srow + n0], silu_f(acc[0][g][r]) * cc);
      atomicAdd(&sums[srow + n1], silu_f(acc[1][g][r]) * cc);
    }
}

// ---------------- node MLP 1 (concat -> 2 layers -> residual) ----------------
__global__ __launch_bounds__(256) void node_mlp1_kernel(
    const float* __restrict__ h, const float* __restrict__ hn32,
    const h16* __restrict__ hnh,
    const float* __restrict__ sums, const float* __restrict__ cnt,
    const h16* __restrict__ Wn1t, const float* __restrict__ bn1,
    const h16* __restrict__ Wn2t, const float* __restrict__ bn2,
    float* __restrict__ h1) {
  __shared__ h16 A1[16 * 392];  // 16 x 384 (K = 128 hn + 256 m_aggr)
  __shared__ h16 M1[16 * 264];
  const int tid  = threadIdx.x;
  const int base = blockIdx.x * 16;
  {
    int m = tid >> 4;
    int c = tid & 15;
    int i = base + m;
    const h16* src = hnh + (size_t)i * 128 + c * 8;
    *(v8h*)&A1[m * 392 + c * 8] = *(const v8h*)src;
    float inv = 1.0f / fmaxf(cnt[i], 1.0f);
    const float* srow = sums + (size_t)i * 256 + c * 16;
#pragma unroll
    for (int j = 0; j < 16; ++j)
      A1[m * 392 + 128 + c * 16 + j] = (h16)(srow[j] * inv);
  }
  __syncthreads();

  const int lane   = tid & 31;
  const int wave   = tid >> 5;
  const int rowoff = (lane & 16) ? 8 : 0;
  const int ncl    = lane & 15;

  {  // layer 1: out 16x256, two N-tiles per wave, shared A frag across tiles
    const int tile0 = wave * 2, tile1 = wave * 2 + 1;
    const int n0 = tile0 * 16 + ncl, n1 = tile1 * 16 + ncl;
    v8f acc0, acc1;
    float b00 = bn1[n0], b01 = bn1[n1];
#pragma unroll
    for (int r = 0; r < 8; ++r) { acc0[r] = b00; acc1[r] = b01; }
#pragma unroll
    for (int kk = 0; kk < 12; ++kk) {
      v16h a  = load_a_frag(A1, 392, lane, kk);
      v16h b0 = load_b_frag(Wn1t, 384, lane, tile0, kk);
      v16h b1 = load_b_frag(Wn1t, 384, lane, tile1, kk);
      acc0 = WMMA_F16(a, b0, acc0);
      acc1 = WMMA_F16(a, b1, acc1);
    }
#pragma unroll
    for (int r = 0; r < 8; ++r) {
      M1[(r + rowoff) * 264 + n0] = (h16)silu_f(acc0[r]);
      M1[(r + rowoff) * 264 + n1] = (h16)silu_f(acc1[r]);
    }
  }
  __syncthreads();

  {  // layer 2: 16x128 out, 8 tiles, one per wave
    int tile = wave;
    int n    = tile * 16 + ncl;
    float b0 = bn2[n];
    v8f acc;
#pragma unroll
    for (int r = 0; r < 8; ++r) acc[r] = b0;
#pragma unroll
    for (int kk = 0; kk < 8; ++kk) {
      v16h a = load_a_frag(M1, 264, lane, kk);
      v16h b = load_b_frag(Wn2t, 256, lane, tile, kk);
      acc = WMMA_F16(a, b, acc);
    }
#pragma unroll
    for (int r = 0; r < 8; ++r) {
      int m = r + rowoff;
      size_t off = (size_t)(base + m) * 128 + n;
      h1[off] = h[off] + hn32[off] + acc[r];  // h1 = h_res + hn + hd
    }
  }
}

// ---------------- node MLP 2 (final FFN + residual) ----------------
__global__ __launch_bounds__(256) void node_mlp2_kernel(
    const h16* __restrict__ h2nh,
    const h16* __restrict__ Wm1t, const float* __restrict__ bm1,
    const h16* __restrict__ Wm2t, const float* __restrict__ bm2,
    const float* __restrict__ h1, float* __restrict__ out) {
  __shared__ h16 A2[16 * 136];  // 16 x 128
  __shared__ h16 M1[16 * 264];
  const int tid  = threadIdx.x;
  const int base = blockIdx.x * 16;
  {
    int m = tid >> 4;
    int c = tid & 15;
    int i = base + m;
    const h16* src = h2nh + (size_t)i * 128 + c * 8;
    *(v8h*)&A2[m * 136 + c * 8] = *(const v8h*)src;
  }
  __syncthreads();

  const int lane   = tid & 31;
  const int wave   = tid >> 5;
  const int rowoff = (lane & 16) ? 8 : 0;
  const int ncl    = lane & 15;

  {  // layer 1: out 16x256, two N-tiles per wave, shared A frag
    const int tile0 = wave * 2, tile1 = wave * 2 + 1;
    const int n0 = tile0 * 16 + ncl, n1 = tile1 * 16 + ncl;
    v8f acc0, acc1;
    float b00 = bm1[n0], b01 = bm1[n1];
#pragma unroll
    for (int r = 0; r < 8; ++r) { acc0[r] = b00; acc1[r] = b01; }
#pragma unroll
    for (int kk = 0; kk < 4; ++kk) {
      v16h a  = load_a_frag(A2, 136, lane, kk);
      v16h b0 = load_b_frag(Wm1t, 128, lane, tile0, kk);
      v16h b1 = load_b_frag(Wm1t, 128, lane, tile1, kk);
      acc0 = WMMA_F16(a, b0, acc0);
      acc1 = WMMA_F16(a, b1, acc1);
    }
#pragma unroll
    for (int r = 0; r < 8; ++r) {
      M1[(r + rowoff) * 264 + n0] = (h16)silu_f(acc0[r]);
      M1[(r + rowoff) * 264 + n1] = (h16)silu_f(acc1[r]);
    }
  }
  __syncthreads();

  {
    int tile = wave;
    int n    = tile * 16 + ncl;
    float b0 = bm2[n];
    v8f acc;
#pragma unroll
    for (int r = 0; r < 8; ++r) acc[r] = b0;
#pragma unroll
    for (int kk = 0; kk < 8; ++kk) {
      v16h a = load_a_frag(M1, 264, lane, kk);
      v16h b = load_b_frag(Wm2t, 256, lane, tile, kk);
      acc = WMMA_F16(a, b, acc);
    }
#pragma unroll
    for (int r = 0; r < 8; ++r) {
      int m = r + rowoff;
      size_t off = (size_t)(base + m) * 128 + n;
      out[off] = h1[off] + acc[r];
    }
  }
}

// ---------------- host launch ----------------

extern "C" void kernel_launch(void* const* d_in, const int* in_sizes, int n_in,
                              void* d_out, int out_size, void* d_ws, size_t ws_size,
                              hipStream_t stream) {
  const float* x   = (const float*)d_in[0];
  const float* h   = (const float*)d_in[1];
  const int*   ei  = (const int*)d_in[2];
  const float* We1 = (const float*)d_in[3];
  const float* be1 = (const float*)d_in[4];
  const float* We2 = (const float*)d_in[5];
  const float* be2 = (const float*)d_in[6];
  const float* Wn1 = (const float*)d_in[7];
  const float* bn1 = (const float*)d_in[8];
  const float* Wn2 = (const float*)d_in[9];
  const float* bn2 = (const float*)d_in[10];
  const float* Wm1 = (const float*)d_in[11];
  const float* bm1 = (const float*)d_in[12];
  const float* Wm2 = (const float*)d_in[13];
  const float* bm2 = (const float*)d_in[14];
  const float* g1  = (const float*)d_in[15];
  const float* bt1 = (const float*)d_in[16];
  const float* g2  = (const float*)d_in[17];
  const float* bt2 = (const float*)d_in[18];

  const int N = in_sizes[1] / 128;  // 10000
  const int E = in_sizes[2] / 2;    // 320000

  // workspace carve-out (256B aligned slices)
  char* ws = (char*)d_ws;
  size_t off = 0;
  auto take = [&](size_t bytes) {
    off = (off + 255) & ~(size_t)255;
    char* p = ws + off;
    off += bytes;
    return p;
  };
  float* sums  = (float*)take((size_t)N * 257 * sizeof(float));  // sums then cnt contiguous
  float* cnt   = sums + (size_t)N * 256;
  float* hn32  = (float*)take((size_t)N * 128 * sizeof(float));
  h16*   hnh   = (h16*)  take((size_t)N * 128 * sizeof(h16));
  float* h1    = (float*)take((size_t)N * 128 * sizeof(float));
  h16*   h2nh  = (h16*)  take((size_t)N * 128 * sizeof(h16));
  h16*   We1t  = (h16*)  take((size_t)256 * 256 * sizeof(h16));
  h16*   We2t  = (h16*)  take((size_t)256 * 256 * sizeof(h16));
  h16*   Wn1t  = (h16*)  take((size_t)256 * 384 * sizeof(h16));
  h16*   Wn2t  = (h16*)  take((size_t)128 * 256 * sizeof(h16));
  h16*   Wm1t  = (h16*)  take((size_t)256 * 128 * sizeof(h16));
  h16*   Wm2t  = (h16*)  take((size_t)128 * 256 * sizeof(h16));

  // 1) zero segment-sum accumulators (sums + cnt in one pass)
  {
    int n = N * 257;
    zero_kernel<<<(n + 255) / 256, 256, 0, stream>>>(sums, n);
  }
  // 2) LN #1: hn (fp32 + f16)
  ln_kernel<<<N, 128, 0, stream>>>(h, g1, bt1, hn32, hnh);

  // 3) weight transposes into N-major f16 (B-fragment friendly)
  auto tr = [&](const float* src, h16* dst, int K, int Nw) {
    int n = K * Nw;
    transpose_f16_kernel<<<(n + 255) / 256, 256, 0, stream>>>(src, dst, K, Nw);
  };
  tr(We1, We1t, 256, 256);   // first 256 rows; row 256 (dist) handled in fp32
  tr(We2, We2t, 256, 256);
  tr(Wn1, Wn1t, 384, 256);
  tr(Wn2, Wn2t, 256, 128);
  tr(Wm1, Wm1t, 128, 256);
  tr(Wm2, Wm2t, 256, 128);

  // 4) fused edge MLP + cutoff + scatter (dominant: ~84 GFLOP via WMMA f16)
  edge_mlp_kernel<<<E / 64, 256, 0, stream>>>(
      x, ei, E, hnh, We1t, We1 + (size_t)256 * 256, be1, We2t, be2, sums, cnt);

  // 5) node MLP 1 + residuals -> h1
  node_mlp1_kernel<<<N / 16, 256, 0, stream>>>(
      h, hn32, hnh, sums, cnt, Wn1t, bn1, Wn2t, bn2, h1);

  // 6) LN #2 -> f16
  ln_kernel<<<N, 128, 0, stream>>>(h1, g2, bt2, nullptr, h2nh);

  // 7) final FFN + residual -> out
  node_mlp2_kernel<<<N / 16, 256, 0, stream>>>(
      h2nh, Wm1t, bm1, Wm2t, bm2, h1, (float*)d_out);

  (void)n_in; (void)out_size; (void)ws_size;
}
